// LocationSensitiveAttention_40759239639196
// MI455X (gfx1250) — matrix operand order repaired and checked by reading, not compiled, not run on hardware
//
#include <hip/hip_runtime.h>
#include <stdint.h>

// Problem constants (from reference)
#define B_DIM 64
#define T_DIM 2048
#define ENC   512
#define QD    256
#define AD    128
#define CCH   32
#define KSZ   31
#define PADW  15

// Tiling
#define TILE_T  128            // M rows per workgroup
#define KSTEPS  (ENC / 32)     // 16 k-steps of K=32
#define FSTRIDE 36             // floats per LDS row (32 used + 4 pad; 144 B, 16B-aligned)

typedef _Float16 v16h __attribute__((ext_vector_type(16)));
typedef float    v8f  __attribute__((ext_vector_type(8)));
typedef unsigned int u32x4 __attribute__((ext_vector_type(4)));
typedef int          i32x8 __attribute__((ext_vector_type(8)));
typedef int          i32x4 __attribute__((ext_vector_type(4)));

static __device__ __forceinline__ uint32_t pack2h(float a, float b) {
  union { _Float16 h[2]; uint32_t u; } p;
  p.h[0] = (_Float16)a; p.h[1] = (_Float16)b;
  return p.u;
}

// A-fragment (16x32 f16) built from an fp32 LDS tile.
// lane L<16 -> row M=L, halves K=0..7 & 16..23 ; lane L>=16 -> row M=L-16, K=8..15 & 24..31.
static __device__ __forceinline__ v16h read_frag_f32lds(const float* tile_base, int lane) {
  const float* p = tile_base + (lane & 15) * FSTRIDE + ((lane >> 4) << 3);
  float4 x0 = *(const float4*)(p);
  float4 x1 = *(const float4*)(p + 4);
  float4 x2 = *(const float4*)(p + 16);
  float4 x3 = *(const float4*)(p + 20);
  union { uint32_t u[8]; v16h v; } r;
  r.u[0] = pack2h(x0.x, x0.y); r.u[1] = pack2h(x0.z, x0.w);
  r.u[2] = pack2h(x1.x, x1.y); r.u[3] = pack2h(x1.z, x1.w);
  r.u[4] = pack2h(x2.x, x2.y); r.u[5] = pack2h(x2.z, x2.w);
  r.u[6] = pack2h(x3.x, x3.y); r.u[7] = pack2h(x3.z, x3.w);
  return r.v;
}

// B-fragment (32x16 f16) from fragment-packed global memory: 8 dwords per lane.
static __device__ __forceinline__ v16h read_frag_g(const uint32_t* __restrict__ g) {
  union { uint4 q[2]; v16h v; } u;
  u.q[0] = *(const uint4*)(g);
  u.q[1] = *(const uint4*)(g + 4);
  return u.v;
}

// ---- TDM: one descriptor-driven DMA of a 128x32 fp32 tile (row-major, tensor row
// stride 512 floats) into LDS with +4-dword padding after every 32 dwords, so rows
// land at FSTRIDE(36)-float pitch. D# layout per CDNA5 ISA 8.3/8.4.
// NOTE: this toolchain's builtin takes 6 args (g0, g1, g2, g3, g4, cpol).
static __device__ __forceinline__ void tdm_load_slab(uint32_t lds_addr,
                                                     unsigned long long gaddr) {
  u32x4 g0;
  g0.x = 1u;                                        // count=1 (valid), load, no gather
  g0.y = lds_addr;                                  // LDS byte address
  g0.z = (uint32_t)gaddr;                           // global_addr[31:0]
  g0.w = (uint32_t)(gaddr >> 32) | (2u << 30);      // global_addr[56:32] | type=2
  i32x8 g1;
  g1[0] = (2 << 16)        // data_size = 4 bytes
        | (1 << 20)        // pad_enable
        | (4 << 22)        // pad_interval: 32 dwords
        | (3 << 25);       // pad_amount: 4 dwords
  g1[1] = (int)(512u << 16);          // tensor_dim0 = 512 (low 16 bits)
  g1[2] = 0;                          // tensor_dim0 hi, tensor_dim1 low16 (131072 -> 0)
  g1[3] = 2 | (32 << 16);             // tensor_dim1 hi (131072>>16=2) | tile_dim0 = 32
  g1[4] = 128;                        // tile_dim1 = 128, tile_dim2 = 0
  g1[5] = 512;                        // tensor_dim0_stride = 512 elements
  g1[6] = 0;
  g1[7] = 0;
  i32x4 gz  = {0, 0, 0, 0};           // groups 2/3 unused (2D tensor)
  i32x8 gz8 = {0, 0, 0, 0, 0, 0, 0, 0};
  __builtin_amdgcn_tensor_load_to_lds(g0, g1, gz, gz, gz8, 0);
}

static __device__ __forceinline__ void async_wait0() {
  asm volatile("s_wait_asynccnt 0x0" ::: "memory");
}

// ---- Prep: pack Wk [512x128] f32 -> f16 fragment-major: [nt(8)][ks(16)][lane(32)][j(8)] dwords.
__global__ __launch_bounds__(256) void pack_wk_kernel(const float* __restrict__ Wk,
                                                      uint32_t* __restrict__ out) {
  int idx  = blockIdx.x * 256 + threadIdx.x;     // 32768 total
  int j    = idx & 7;
  int lane = (idx >> 3) & 31;
  int ks   = (idx >> 8) & 15;
  int nt   = idx >> 12;
  int k0   = ks * 32 + ((j >> 2) << 4) + ((lane >> 4) << 3) + 2 * (j & 3);
  int n    = nt * 16 + (lane & 15);
  out[idx] = pack2h(Wk[k0 * AD + n], Wk[(k0 + 1) * AD + n]);
}

// ---- Prep: pack Wl [32x128] the same way (single k-step, K=32).
__global__ __launch_bounds__(256) void pack_wl_kernel(const float* __restrict__ Wl,
                                                      uint32_t* __restrict__ out) {
  int idx  = blockIdx.x * 256 + threadIdx.x;     // 2048 total
  int j    = idx & 7;
  int lane = (idx >> 3) & 31;
  int nt   = idx >> 8;
  int k0   = ((j >> 2) << 4) + ((lane >> 4) << 3) + 2 * (j & 3);
  int n    = nt * 16 + (lane & 15);
  out[idx] = pack2h(Wl[k0 * AD + n], Wl[(k0 + 1) * AD + n]);
}

// ---- Prep: query = decoder_hidden @ Wq  (64x256 @ 256x128 -> tiny, scalar)
__global__ __launch_bounds__(256) void query_kernel(const float* __restrict__ dh,
                                                    const float* __restrict__ Wq,
                                                    float* __restrict__ q) {
  int idx = blockIdx.x * 256 + threadIdx.x;      // 8192 total
  int b = idx >> 7, a = idx & (AD - 1);
  float s = 0.f;
  for (int k = 0; k < QD; ++k) s += dh[b * QD + k] * Wq[k * AD + a];
  q[idx] = s;
}

// ---- Main fused kernel: keys GEMM + location conv + loc GEMM + query + tanh + v-dot.
__global__ __launch_bounds__(256) void energy_kernel(
    const float* __restrict__ enc, const float* __restrict__ patt,
    const float* __restrict__ convw, const float* __restrict__ convb,
    const uint32_t* __restrict__ pWk, const uint32_t* __restrict__ pWl,
    const float* __restrict__ query, const float* __restrict__ vvec,
    float* __restrict__ energy) {
  __shared__ float aLds[2][TILE_T * FSTRIDE];   // fp32 A slabs (TDM double-buffered)
  __shared__ float convwLds[CCH * KSZ];         // 992 floats, async-staged
  __shared__ float energyLds[TILE_T];
  __shared__ float pattLds[TILE_T + 2 * PADW + 2];

  const int tid  = threadIdx.x;
  const int lane = tid & 31;
  const int w    = tid >> 5;      // 8 waves
  const int wm   = w & 3;         // wave M group: rows [wm*32, wm*32+32)
  const int wn   = w >> 2;        // wave N group: cols [wn*64, wn*64+64)
  const int bb   = blockIdx.y;
  const int t0   = blockIdx.x * TILE_T;

  // Async-stage conv weights (3968 B = 248 x 16 B) into LDS; consumed much later.
  {
    const uint32_t cwBase = (uint32_t)(uintptr_t)&convwLds[0];
    if (tid < (CCH * KSZ * 4) / 16) {
      uint32_t ldsoff = cwBase + (uint32_t)tid * 16u;
      uint32_t voff   = (uint32_t)tid * 16u;
      asm volatile("global_load_async_to_lds_b128 %0, %1, %2 offset:0"
                   :: "v"(ldsoff), "v"(voff),
                      "s"((unsigned long long)(uintptr_t)convw)
                   : "memory");
    }
  }

  const v8f zero = {};
  v8f acc[2][4];
#pragma unroll
  for (int m = 0; m < 2; ++m)
#pragma unroll
    for (int j = 0; j < 4; ++j) acc[m][j] = zero;

  const float* gslab0 = enc + ((size_t)bb * T_DIM + t0) * ENC;
  const uint32_t ldsB0 = (uint32_t)(uintptr_t)&aLds[0][0];
  const uint32_t ldsB1 = (uint32_t)(uintptr_t)&aLds[1][0];

  // Prologue: TDM-stage k-step 0 (wave 0 issues; TENSORcnt tracks completion).
  if (w == 0) {
    tdm_load_slab(ldsB0, (unsigned long long)(uintptr_t)gslab0);
    __builtin_amdgcn_s_wait_tensorcnt(0);
  }
  __syncthreads();

  int cur = 0;
  for (int ks = 0; ks < KSTEPS; ++ks) {
    if (ks + 1 < KSTEPS && w == 0)   // kick off next slab into the other buffer
      tdm_load_slab(cur ? ldsB0 : ldsB1,
                    (unsigned long long)(uintptr_t)(gslab0 + (ks + 1) * 32));

    v16h a0 = read_frag_f32lds(&aLds[cur][(wm * 32) * FSTRIDE], lane);
    v16h a1 = read_frag_f32lds(&aLds[cur][(wm * 32 + 16) * FSTRIDE], lane);
#pragma unroll
    for (int j = 0; j < 4; ++j) {
      v16h bf = read_frag_g(pWk + ((((wn * 4 + j) * KSTEPS + ks) * 32 + lane) << 3));
      acc[0][j] = __builtin_amdgcn_wmma_f32_16x16x32_f16(false, a0, false, bf,
                                                         (short)0, acc[0][j], false, false);
      acc[1][j] = __builtin_amdgcn_wmma_f32_16x16x32_f16(false, a1, false, bf,
                                                         (short)0, acc[1][j], false, false);
    }
    if (ks + 1 < KSTEPS && w == 0)
      __builtin_amdgcn_s_wait_tensorcnt(0);   // next buffer fully landed
    __syncthreads();                          // all readers done + DMA visible
    cur ^= 1;
  }

  // ---- Location term: conv1d(prev_attention) tile -> fp32 [128 x 32] -> one K=32 WMMA round.
  if (tid < TILE_T + 2 * PADW) {
    int gi = t0 - PADW + tid;
    pattLds[tid] = (gi >= 0 && gi < T_DIM) ? patt[(size_t)bb * T_DIM + gi] : 0.f;
  }
  if (tid < TILE_T) energyLds[tid] = 0.f;
  async_wait0();                              // conv weights are in LDS by now
  __syncthreads();

  for (int e = tid; e < TILE_T * CCH; e += 256) {
    int tl = e >> 5, c = e & 31;
    float s = convb[c];
#pragma unroll
    for (int j = 0; j < KSZ; ++j) s += convwLds[c * KSZ + j] * pattLds[tl + j];
    aLds[0][tl * FSTRIDE + c] = s;            // reuse buffer 0 as loc tile
  }
  __syncthreads();

  {
    v16h a0 = read_frag_f32lds(&aLds[0][(wm * 32) * FSTRIDE], lane);
    v16h a1 = read_frag_f32lds(&aLds[0][(wm * 32 + 16) * FSTRIDE], lane);
#pragma unroll
    for (int j = 0; j < 4; ++j) {
      v16h bf = read_frag_g(pWl + (((wn * 4 + j) * 32 + lane) << 3));
      acc[0][j] = __builtin_amdgcn_wmma_f32_16x16x32_f16(false, a0, false, bf,
                                                         (short)0, acc[0][j], false, false);
      acc[1][j] = __builtin_amdgcn_wmma_f32_16x16x32_f16(false, a1, false, bf,
                                                         (short)0, acc[1][j], false, false);
    }
  }

  // ---- Epilogue: + query, tanh, dot with v, reduce across N via LDS float atomics.
  // C/D layout: VGPR r, lanes 0-15 -> M=r, lanes 16-31 -> M=r+8; N = lane&15.
  float qv[4], vv[4];
#pragma unroll
  for (int j = 0; j < 4; ++j) {
    int a = (wn * 4 + j) * 16 + (lane & 15);
    qv[j] = query[bb * AD + a];
    vv[j] = vvec[a];
  }
#pragma unroll
  for (int m = 0; m < 2; ++m) {
    int tb = wm * 32 + m * 16 + ((lane >> 4) << 3);
#pragma unroll
    for (int r = 0; r < 8; ++r) {
      float p = 0.f;
#pragma unroll
      for (int j = 0; j < 4; ++j) p += tanhf(acc[m][j][r] + qv[j]) * vv[j];
      atomicAdd(&energyLds[tb + r], p);   // ds_add_f32
    }
  }
  __syncthreads();
  if (tid < TILE_T) energy[(size_t)bb * T_DIM + t0 + tid] = energyLds[tid];
}

// ---- Softmax over T per batch row.
__global__ __launch_bounds__(256) void softmax_kernel(const float* __restrict__ energy,
                                                      float* __restrict__ out) {
  __shared__ float red[256];
  const int b = blockIdx.x, tid = threadIdx.x;
  const float* e = energy + (size_t)b * T_DIM;
  float ev[8];
  float mx = -3.402823466e+38f;
#pragma unroll
  for (int i = 0; i < 8; ++i) { ev[i] = e[tid + i * 256]; mx = fmaxf(mx, ev[i]); }
  red[tid] = mx; __syncthreads();
  for (int s = 128; s > 0; s >>= 1) {
    if (tid < s) red[tid] = fmaxf(red[tid], red[tid + s]);
    __syncthreads();
  }
  mx = red[0]; __syncthreads();
  float sum = 0.f;
#pragma unroll
  for (int i = 0; i < 8; ++i) { ev[i] = __expf(ev[i] - mx); sum += ev[i]; }
  red[tid] = sum; __syncthreads();
  for (int s = 128; s > 0; s >>= 1) {
    if (tid < s) red[tid] += red[tid + s];
    __syncthreads();
  }
  float inv = 1.0f / red[0];
#pragma unroll
  for (int i = 0; i < 8; ++i) out[(size_t)b * T_DIM + tid + i * 256] = ev[i] * inv;
}

extern "C" void kernel_launch(void* const* d_in, const int* in_sizes, int n_in,
                              void* d_out, int out_size, void* d_ws, size_t ws_size,
                              hipStream_t stream) {
  (void)in_sizes; (void)n_in; (void)out_size; (void)ws_size;
  const float* enc   = (const float*)d_in[0];  // [64,2048,512]
  const float* dh    = (const float*)d_in[1];  // [64,256]
  const float* patt  = (const float*)d_in[2];  // [64,2048]
  const float* Wq    = (const float*)d_in[3];  // [256,128]
  const float* Wk    = (const float*)d_in[4];  // [512,128]
  const float* convw = (const float*)d_in[5];  // [32,1,31]
  const float* convb = (const float*)d_in[6];  // [32]
  const float* Wl    = (const float*)d_in[7];  // [32,128]
  const float* v     = (const float*)d_in[8];  // [128]
  float* out = (float*)d_out;                  // [64,2048]

  char* ws = (char*)d_ws;
  uint32_t* pWk = (uint32_t*)(ws + 0);         // 131072 B: packed f16 Wk fragments
  uint32_t* pWl = (uint32_t*)(ws + 131072);    //   8192 B: packed f16 Wl fragments
  float*    q   = (float*)   (ws + 139264);    //  32768 B: query [64,128]
  float*    egy = (float*)   (ws + 172032);    // 524288 B: energy [64,2048]

  pack_wk_kernel<<<128, 256, 0, stream>>>(Wk, pWk);
  pack_wl_kernel<<<8, 256, 0, stream>>>(Wl, pWl);
  query_kernel<<<32, 256, 0, stream>>>(dh, Wq, q);
  dim3 grid(T_DIM / TILE_T, B_DIM);
  energy_kernel<<<grid, 256, 0, stream>>>(enc, patt, convw, convb, pWk, pWl, q, v, egy);
  softmax_kernel<<<B_DIM, 256, 0, stream>>>(egy, out);
}
